// PFASModule_86749749444952
// MI455X (gfx1250) — matrix operand order repaired and checked by reading, not compiled, not run on hardware
//
#include <hip/hip_runtime.h>
#include <hip/hip_bf16.h>

#define NPTS  16384
#define NBP   8192      // points per batch
#define KNN   16
#define CHUNK 1024
#define CIN   256
#define HID   64

typedef __attribute__((ext_vector_type(16))) _Float16 v16h;
typedef __attribute__((ext_vector_type(8)))  float    v8f;

// ---------------------------------------------------------------------------
// Kernel 1: fused KNN (chunk-masked top-16) + density + covariance eigenvalues
// One block = 256 queries (one lane per query). Whole batch's coords (96 KB)
// staged in LDS (CDNA5 WGP has 320 KB) -> candidate loop reads are LDS
// broadcasts; neighbor gather for covariance also hits LDS.
// ---------------------------------------------------------------------------
extern "C" __global__ __launch_bounds__(256)
void pfas_knn_geom(const float* __restrict__ coord,
                   float* __restrict__ lin,
                   float* __restrict__ den)
{
    extern __shared__ float smem[];          // 3 * NBP floats = 96 KB (SoA)
    float* sx = smem;
    float* sy = smem + NBP;
    float* sz = smem + 2 * NBP;

    const int batch = blockIdx.x >> 5;       // 32 blocks per batch
    const int blk   = blockIdx.x & 31;
    const int tid   = threadIdx.x;

    for (int j = tid; j < NBP; j += 256) {
        const float* c = coord + (size_t)(batch * NBP + j) * 3;
        sx[j] = c[0]; sy[j] = c[1]; sz[j] = c[2];
    }
    __syncthreads();

    const int   qlocal = blk * 256 + tid;    // 256 <= CHUNK: block-uniform chunk
    const int   qg     = batch * NBP + qlocal;
    const float qx = sx[qlocal], qy = sy[qlocal], qz = sz[qlocal];
    const int   qchunk = qlocal >> 10;

    float bd[KNN]; int bi[KNN];
    #pragma unroll
    for (int t = 0; t < KNN; ++t) { bd[t] = 3.0e38f; bi[t] = 0; }
    float worst = 3.0e38f;

    for (int j = 0; j < NBP; ++j) {
        if ((j >> 10) == qchunk) { j += CHUNK - 1; continue; }   // scalar-uniform skip
        float dx = sx[j] - qx, dy = sy[j] - qy, dz = sz[j] - qz;
        float d2 = fmaf(dx, dx, fmaf(dy, dy, dz * dz));
        if (d2 < worst) {                    // rare after warm-up -> s_cbranch_execz
            float cd = d2; int ci = j;
            #pragma unroll
            for (int t = 0; t < KNN; ++t) {  // stable sorted insert (strict <)
                if (cd < bd[t]) {
                    float td = bd[t]; int ti = bi[t];
                    bd[t] = cd; bi[t] = ci; cd = td; ci = ti;
                }
            }
            worst = bd[KNN - 1];
        }
    }

    // density = 1 / (mean neighbor distance + 1e-6)
    float sd = 0.f;
    #pragma unroll
    for (int t = 0; t < KNN; ++t) sd += sqrtf(fmaxf(bd[t], 1e-12f));
    const float density = 1.0f / (sd * (1.0f / KNN) + 1e-6f);

    // covariance of neighbor coords (two LDS-gather passes, no big reg arrays)
    float mx = 0.f, my = 0.f, mz = 0.f;
    #pragma unroll
    for (int t = 0; t < KNN; ++t) { mx += sx[bi[t]]; my += sy[bi[t]]; mz += sz[bi[t]]; }
    mx *= (1.0f / KNN); my *= (1.0f / KNN); mz *= (1.0f / KNN);

    float cxx = 0.f, cxy = 0.f, cxz = 0.f, cyy = 0.f, cyz = 0.f, czz = 0.f;
    #pragma unroll
    for (int t = 0; t < KNN; ++t) {
        float ax = sx[bi[t]] - mx, ay = sy[bi[t]] - my, az = sz[bi[t]] - mz;
        cxx = fmaf(ax, ax, cxx); cxy = fmaf(ax, ay, cxy); cxz = fmaf(ax, az, cxz);
        cyy = fmaf(ay, ay, cyy); cyz = fmaf(ay, az, cyz); czz = fmaf(az, az, czz);
    }
    const float inv = 1.0f / (KNN - 1);
    cxx *= inv; cxy *= inv; cxz *= inv; cyy *= inv; cyz *= inv; czz *= inv;

    // closed-form eigenvalues of symmetric PSD 3x3 (descending)
    float q  = (cxx + cyy + czz) * (1.0f / 3.0f);
    float p1 = cxy * cxy + cxz * cxz + cyz * cyz;
    float bxx = cxx - q, byy = cyy - q, bzz = czz - q;
    float p2 = bxx * bxx + byy * byy + bzz * bzz + 2.0f * p1;
    float e1, e2, e3;
    if (p2 < 1e-24f) {
        e1 = e2 = e3 = q;
    } else {
        float p  = sqrtf(p2 * (1.0f / 6.0f));
        float ip = 1.0f / p;
        float detB = (bxx * (byy * bzz - cyz * cyz)
                    - cxy * (cxy * bzz - cyz * cxz)
                    + cxz * (cxy * cyz - byy * cxz)) * (ip * ip * ip);
        float r = fminf(1.0f, fmaxf(-1.0f, 0.5f * detB));
        float phi = acosf(r) * (1.0f / 3.0f);
        e1 = q + 2.0f * p * __cosf(phi);
        e3 = q + 2.0f * p * __cosf(phi + 2.09439510239f);   // + 2*pi/3
        e2 = 3.0f * q - e1 - e3;
    }
    const float tr = cxx + cyy + czz;        // == sum of eigenvalues
    lin[qg] = (e1 - e2 - e3) / (tr + 1e-6f); // Sn0 - (Sn1 + Sn2)
    den[qg] = density;
}

// ---------------------------------------------------------------------------
// Kernel 2: h = feat @ W1 + b1 via v_wmma_f32_16x16x32_f16 (f32 accumulate).
// Block = 128 thr = 4 waves; block owns 16 rows, each wave one 16-col tile.
// A/B VGPR layouts: contiguous K, lane half h holds K 16h..16h+15 (16-bit 16x32 A; 32x16 B).
// ---------------------------------------------------------------------------
extern "C" __global__ __launch_bounds__(128)
void pfas_gemm1(const float* __restrict__ feat,
                const float* __restrict__ W1,
                const float* __restrict__ b1,
                float* __restrict__ h)
{
    const int wave = threadIdx.x >> 5;
    const int lane = threadIdx.x & 31;
    const int lh   = lane & 15;
    const int hi   = lane >> 4;
    const int row0 = blockIdx.x * 16;
    const int col0 = wave * 16;

    const float* fr   = feat + (size_t)(row0 + lh) * CIN;  // A row for this lane
    const int    colB = col0 + lh;                         // B column for this lane
    const int    lo   = hi * 16;                           // A: K half select
    const int    kOff = hi * 16;                           // B: K half select

    v8f c = {};
    #pragma unroll
    for (int it = 0; it < CIN / 32; ++it) {
        const int kb = it * 32;
        v16h a, b;
        #pragma unroll
        for (int j = 0; j < 8; ++j) {        // A 16x32: elems 0-7 = K lo..lo+7,
            a[j]     = (_Float16)fr[kb + lo + j];          // elems 8-15 = K lo+8..
            a[j + 8] = (_Float16)fr[kb + lo + 8 + j];
        }
        #pragma unroll
        for (int j = 0; j < 16; ++j)         // B 32x16: lane half selects K range
            b[j] = (_Float16)W1[(size_t)(kb + kOff + j) * HID + colB];
        if (it + 1 < CIN / 32)
            __builtin_prefetch(fr + kb + 32, 0, 1);        // global_prefetch_b8
        c = __builtin_amdgcn_wmma_f32_16x16x32_f16(
                /*neg_a=*/false, a, /*neg_b=*/false, b,
                /*c_mod=*/(short)0, c, /*reuse_a=*/false, /*reuse_b=*/false);
    }

    const float bias = b1[colB];
    #pragma unroll
    for (int r = 0; r < 8; ++r) {            // C: vgpr r -> M = r + 8*hi, N = lh
        int row = row0 + r + hi * 8;
        h[(size_t)row * HID + colB] = c[r] + bias;
    }
}

// ---------------------------------------------------------------------------
// Kernel 3: per-(batch, channel) BN statistics (biased var, as torch BN)
// ---------------------------------------------------------------------------
extern "C" __global__ __launch_bounds__(256)
void pfas_bnstats(const float* __restrict__ h,
                  float* __restrict__ mu, float* __restrict__ var)
{
    __shared__ float s1[256], s2[256];
    const int b   = blockIdx.x >> 6;
    const int ch  = blockIdx.x & 63;
    const int tid = threadIdx.x;
    float a1 = 0.f, a2 = 0.f;
    for (int r = tid; r < NBP; r += 256) {
        float v = h[(size_t)(b * NBP + r) * HID + ch];
        a1 += v; a2 = fmaf(v, v, a2);
    }
    s1[tid] = a1; s2[tid] = a2;
    __syncthreads();
    for (int s = 128; s > 0; s >>= 1) {
        if (tid < s) { s1[tid] += s1[tid + s]; s2[tid] += s2[tid + s]; }
        __syncthreads();
    }
    if (tid == 0) {
        float m = s1[0] * (1.0f / NBP);
        mu[blockIdx.x]  = m;
        var[blockIdx.x] = s2[0] * (1.0f / NBP) - m * m;
    }
}

// ---------------------------------------------------------------------------
// Kernel 4: BN apply + ReLU + 64->3 GEMV + softmax + grid blend
// ---------------------------------------------------------------------------
extern "C" __global__ __launch_bounds__(256)
void pfas_final(const float* __restrict__ h,
                const float* __restrict__ mu,  const float* __restrict__ var,
                const float* __restrict__ gamma, const float* __restrict__ beta,
                const float* __restrict__ W2,  const float* __restrict__ b2,
                const float* __restrict__ lin, const float* __restrict__ den,
                float* __restrict__ out)
{
    const int i = blockIdx.x * 256 + threadIdx.x;
    if (i >= NPTS) return;
    const int b = i >> 13;                   // i / NBP
    const float* hr   = h   + (size_t)i * HID;
    const float* mub  = mu  + b * HID;
    const float* varb = var + b * HID;

    float l0 = b2[0], l1 = b2[1], l2 = b2[2];
    #pragma unroll 8
    for (int ch = 0; ch < HID; ++ch) {
        float v = (hr[ch] - mub[ch]) * rsqrtf(varb[ch] + 1e-5f) * gamma[ch] + beta[ch];
        v = fmaxf(v, 0.f);
        l0 = fmaf(v, W2[ch * 3 + 0], l0);
        l1 = fmaf(v, W2[ch * 3 + 1], l1);
        l2 = fmaf(v, W2[ch * 3 + 2], l2);
    }
    float m  = fmaxf(l0, fmaxf(l1, l2));
    float e0 = __expf(l0 - m), e1 = __expf(l1 - m), e2 = __expf(l2 - m);
    float is = 1.0f / (e0 + e1 + e2);
    float p0 = e0 * is, p1 = e1 * is, p2 = e2 * is;

    float L = lin[i], D = den[i];
    float tp = (D * 2.0f + p0) * (1.0f / 3.0f);
    float bp = (fmaxf(1.0f - L, 1.0f - D) + p1) * (1.0f / 3.0f);
    float lp = (L * 2.0f + p2) * (1.0f / 3.0f);
    // tower (0.1,0.1,0.1)  background (0.5,0.5,0.5)  line (0.2,0.2,0.05*5)
    out[i * 3 + 0] = tp * 0.1f + bp * 0.5f + lp * 0.20f + 1e-6f;
    out[i * 3 + 1] = tp * 0.1f + bp * 0.5f + lp * 0.20f + 1e-6f;
    out[i * 3 + 2] = tp * 0.1f + bp * 0.5f + lp * 0.25f + 1e-6f;
}

// ---------------------------------------------------------------------------
extern "C" void kernel_launch(void* const* d_in, const int* in_sizes, int n_in,
                              void* d_out, int out_size, void* d_ws, size_t ws_size,
                              hipStream_t stream)
{
    const float* feat  = (const float*)d_in[0];
    const float* coord = (const float*)d_in[1];
    const float* W1    = (const float*)d_in[2];
    const float* b1    = (const float*)d_in[3];
    const float* gamma = (const float*)d_in[4];
    const float* beta  = (const float*)d_in[5];
    const float* W2    = (const float*)d_in[6];
    const float* b2    = (const float*)d_in[7];
    (void)d_in[8];                           // batch: contiguous equal segments
    (void)in_sizes; (void)n_in; (void)out_size; (void)ws_size;
    float* out = (float*)d_out;

    // workspace layout (floats): h[NPTS*HID] | lin[NPTS] | den[NPTS] | mu[128] | var[128]
    float* h   = (float*)d_ws;
    float* lin = h   + (size_t)NPTS * HID;
    float* den = lin + NPTS;
    float* mu  = den + NPTS;
    float* var = mu  + 2 * HID;

    pfas_knn_geom<<<dim3(64),        dim3(256), 3 * NBP * sizeof(float), stream>>>(coord, lin, den);
    pfas_gemm1  <<<dim3(NPTS / 16),  dim3(128), 0, stream>>>(feat, W1, b1, h);
    pfas_bnstats<<<dim3(2 * HID),    dim3(256), 0, stream>>>(h, mu, var);
    pfas_final  <<<dim3(NPTS / 256), dim3(256), 0, stream>>>(h, mu, var, gamma, beta,
                                                             W2, b2, lin, den, out);
}
